// TorchDiAcyclicGraph_33036888440929
// MI455X (gfx1250) — matrix operand order repaired and checked by reading, not compiled, not run on hardware
//
#include <hip/hip_runtime.h>
#include <hip/hip_bf16.h>

// ---------------------------------------------------------------------------
// 16-layer DAG of dense layers: y_i = tanh((y_{i-1}+y_{i-2}) @ W_i + b_i)
// bf16 WMMA (v_wmma_f32_16x16x32_bf16) with f32 accumulation on gfx1250.
// Both GEMM operands staged through LDS; B pre-packed into WMMA lane layout.
// ---------------------------------------------------------------------------

typedef __attribute__((ext_vector_type(16))) __bf16 bf16x16_t;
typedef __attribute__((ext_vector_type(8)))  __bf16 bf16x8_t;
typedef __attribute__((ext_vector_type(8)))  float  f32x8_t;

#define DD      2048
#define BROWS   4096
#define NLAYER  16
#define KTILES  (DD / 32)    // 64 K-steps of 32
#define NTILES  (DD / 16)    // 128 N-tiles of 16
#define TILE_ELEMS 512       // 32x16 bf16 per packed B tile (lane-major, 16 bf16/lane)

union FragBF { bf16x16_t v; bf16x8_t h[2]; uint4 u[2]; };

// ---------------------------------------------------------------------------
// Pack W: fp32 [l][k][n] row-major  ->  bf16 WMMA-B-operand tiles.
// Packed layout: tile index ((l*NTILES + ntile)*KTILES + ktile), within tile
// lane-major: lane holds N = ntile*16 + (lane&15), K = ktile*32 + 16*(lane>>4) + e
// for e = 0..15 (contiguous k order, the same order the A fragments use).
// LDS-staged so global reads and writes are both coalesced.
// ---------------------------------------------------------------------------
__global__ __launch_bounds__(256) void pack_w_kernel(const float* __restrict__ W,
                                                     __bf16* __restrict__ Wpk) {
    __shared__ float lds[32][132];            // 32 k-rows x 128 n-cols (+4 pad)
    const int b  = blockIdx.x;                // ((l*64)+kt)*16 + ng
    const int ng = b & 15;
    const int kt = (b >> 4) & 63;
    const int l  = b >> 10;
    const int t  = threadIdx.x;

    { // coalesced load: 32 rows x 512B
        const int row = t >> 3, seg = t & 7;
        const float* src = W + ((size_t)l * DD + (size_t)(kt * 32 + row)) * DD
                             + (size_t)ng * 128 + seg * 16;
        float4 f0 = ((const float4*)src)[0];
        float4 f1 = ((const float4*)src)[1];
        float4 f2 = ((const float4*)src)[2];
        float4 f3 = ((const float4*)src)[3];
        float* d = &lds[row][seg * 16];
        ((float4*)d)[0] = f0; ((float4*)d)[1] = f1;
        ((float4*)d)[2] = f2; ((float4*)d)[3] = f3;
    }
    __syncthreads();
    { // permuted write: each thread emits one lane's 16 bf16 (32B contiguous)
        const int tl = t >> 5;               // local ntile 0..7
        const int lane = t & 31;
        const int h = lane >> 4, nl = tl * 16 + (lane & 15);
        const size_t ntile = (size_t)ng * 8 + tl;
        __bf16* dst = Wpk + ((((size_t)l * NTILES + ntile) * KTILES + kt) << 9)
                          + lane * 16;
        __align__(16) __bf16 tmp[16];
        #pragma unroll
        for (int e = 0; e < 16; ++e)
            tmp[e] = (__bf16)lds[h * 16 + e][nl];
        ((uint4*)dst)[0] = *(const uint4*)&tmp[0];
        ((uint4*)dst)[1] = *(const uint4*)&tmp[8];
    }
}

// ---------------------------------------------------------------------------
// fp32 -> bf16, 8 elements/thread
// ---------------------------------------------------------------------------
__global__ __launch_bounds__(256) void cvt_f32_bf16_kernel(const float* __restrict__ src,
                                                           __bf16* __restrict__ dst) {
    const size_t i = (size_t)blockIdx.x * 256 + threadIdx.x;
    const float4* s = (const float4*)src + i * 2;
    float4 a = s[0], b = s[1];
    __align__(16) __bf16 t[8];
    t[0] = (__bf16)a.x; t[1] = (__bf16)a.y; t[2] = (__bf16)a.z; t[3] = (__bf16)a.w;
    t[4] = (__bf16)b.x; t[5] = (__bf16)b.y; t[6] = (__bf16)b.z; t[7] = (__bf16)b.w;
    *(uint4*)(dst + i * 8) = *(const uint4*)t;
}

// fast tanh: single v_exp (TRANS, co-executes with WMMA) + fast divide
__device__ __forceinline__ float fast_tanh(float x) {
    float t = x + x;
    t = fminf(fmaxf(t, -30.0f), 30.0f);
    float e = __expf(t);
    return __fdividef(e - 1.0f, e + 1.0f);
}

// ---------------------------------------------------------------------------
// One layer: C[4096,2048] = tanh((A1 (+A2)) @ Wl + bias)
// Block 256 thr (8 waves), tile 128x128. Wave tile 32x64 = 2x4 WMMA accums.
// A panel AND B panel double-buffered in LDS (A: 80B padded rows -> conflict-
// free ds_load_b128 fragments; B: lane-major packed tiles -> exact 64-bank
// coverage). Fan-in sum fused into A staging (v_pk_add_bf16).
// ---------------------------------------------------------------------------
__global__ __launch_bounds__(256) void layer_kernel(
    const __bf16* __restrict__ A1, const __bf16* __restrict__ A2,
    const __bf16* __restrict__ Wl, const float* __restrict__ bias,
    __bf16* __restrict__ outb, float* __restrict__ outf)
{
    __shared__ __align__(16) __bf16 ldsA[2][128 * 40];   // 40 bf16 = 80B row stride, 20KB
    __shared__ __align__(16) __bf16 ldsB[2][8 * TILE_ELEMS]; // 8 packed tiles/kstep, 16KB

    const int tid     = threadIdx.x;
    const int lane    = tid & 31;
    const int wave    = tid >> 5;
    const int waveM   = wave & 3;             // 0..3 -> 32-row bands
    const int waveN   = wave >> 2;            // 0..1 -> 64-col bands
    const int rowBase = blockIdx.y * 128;
    const int colBase = blockIdx.x * 128;
    const int laneRow = lane & 15;
    const int hh      = lane >> 4;

    // A staging: thread -> (row, 16-element half)
    const int sRow  = tid >> 1;
    const int sHalf = tid & 1;
    const size_t sSrc = (size_t)(rowBase + sRow) * DD + sHalf * 16;
    const int sDstA = sRow * 40 + sHalf * 16;

    // B staging: thread -> (local ntile, lane chunk of 32B)
    const int sbNtl  = tid >> 5;              // 0..7
    const int sbLane = tid & 31;
    const size_t ntile0 = (size_t)(colBase >> 4);
    const __bf16* sbSrc = Wl + (((ntile0 + sbNtl) * KTILES) << 9) + sbLane * 16;
    const int sDstB = sbNtl * TILE_ELEMS + sbLane * 16;

    f32x8_t acc[2][4];
    #pragma unroll
    for (int mt = 0; mt < 2; ++mt)
        #pragma unroll
        for (int nt = 0; nt < 4; ++nt)
            #pragma unroll
            for (int e = 0; e < 8; ++e) acc[mt][nt][e] = 0.0f;

    uint4 ra0, ra1, rb0, rb1;
    auto stage = [&](int ks) {
        // ---- A (with fused fan-in sum) ----
        const size_t off = sSrc + (size_t)ks * 32;
        const uint4* p1 = (const uint4*)(A1 + off);
        uint4 x0 = p1[0], x1 = p1[1];
        if (A2) {                              // uniform branch
            const uint4* p2 = (const uint4*)(A2 + off);
            uint4 y0 = p2[0], y1 = p2[1];
            __align__(16) __bf16 xa[16], ya[16];
            *(uint4*)&xa[0] = x0; *(uint4*)&xa[8] = x1;
            *(uint4*)&ya[0] = y0; *(uint4*)&ya[8] = y1;
            #pragma unroll
            for (int i = 0; i < 16; ++i)
                xa[i] = (__bf16)((float)xa[i] + (float)ya[i]);
            ra0 = *(const uint4*)&xa[0]; ra1 = *(const uint4*)&xa[8];
        } else { ra0 = x0; ra1 = x1; }
        // ---- B (already in WMMA lane layout) ----
        const uint4* pb = (const uint4*)(sbSrc + (size_t)ks * TILE_ELEMS);
        rb0 = pb[0]; rb1 = pb[1];
    };
    auto commit = [&](int buf) {
        *(uint4*)&ldsA[buf][sDstA]     = ra0;
        *(uint4*)&ldsA[buf][sDstA + 8] = ra1;
        *(uint4*)&ldsB[buf][sDstB]     = rb0;
        *(uint4*)&ldsB[buf][sDstB + 8] = rb1;
    };

    stage(0);
    commit(0);
    __syncthreads();

    for (int ks = 0; ks < KTILES; ++ks) {
        const int  cur  = ks & 1;
        const bool more = (ks + 1) < KTILES;
        if (more) stage(ks + 1);               // global prefetch of next panels

        FragBF bfrag[4];
        #pragma unroll
        for (int nt = 0; nt < 4; ++nt) {
            const __bf16* bp = &ldsB[cur][(waveN * 4 + nt) * TILE_ELEMS + lane * 16];
            bfrag[nt].h[0] = *(const bf16x8_t*)(bp);
            bfrag[nt].h[1] = *(const bf16x8_t*)(bp + 8);
        }

        FragBF afrag[2];
        #pragma unroll
        for (int mt = 0; mt < 2; ++mt) {
            const int row = waveM * 32 + mt * 16 + laneRow;
            afrag[mt].h[0] = *(const bf16x8_t*)&ldsA[cur][row * 40 + hh * 16];       // K 0..7 / 16..23
            afrag[mt].h[1] = *(const bf16x8_t*)&ldsA[cur][row * 40 + hh * 16 + 8];   // K 8..15 / 24..31
        }

        #pragma unroll
        for (int mt = 0; mt < 2; ++mt)
            #pragma unroll
            for (int nt = 0; nt < 4; ++nt)
                acc[mt][nt] = __builtin_amdgcn_wmma_f32_16x16x32_bf16(
                    false, afrag[mt].v, false, bfrag[nt].v,
                    (short)0, acc[mt][nt], false, false);

        if (more) commit(cur ^ 1);
        __syncthreads();
    }

    // epilogue: bias + tanh, write bf16 (ring) or f32 (terminal)
    #pragma unroll
    for (int nt = 0; nt < 4; ++nt) {
        const int col = colBase + waveN * 64 + nt * 16 + laneRow;
        const float bv = bias[col];
        #pragma unroll
        for (int mt = 0; mt < 2; ++mt) {
            #pragma unroll
            for (int r = 0; r < 8; ++r) {
                const int row = rowBase + waveM * 32 + mt * 16 + hh * 8 + r;
                const float v = fast_tanh(acc[mt][nt][r] + bv);
                const size_t idx = (size_t)row * DD + col;
                if (outf) outf[idx] = v;
                else      outb[idx] = (__bf16)v;
            }
        }
    }
}

// ---------------------------------------------------------------------------
extern "C" void kernel_launch(void* const* d_in, const int* in_sizes, int n_in,
                              void* d_out, int out_size, void* d_ws, size_t ws_size,
                              hipStream_t stream) {
    (void)in_sizes; (void)n_in; (void)out_size; (void)ws_size;
    const float* x    = (const float*)d_in[0];
    const float* W    = (const float*)d_in[1];
    const float* bias = (const float*)d_in[2];
    float* out = (float*)d_out;

    char* ws = (char*)d_ws;
    __bf16* Wpk = (__bf16*)ws;                                   // 128 MB packed bf16 weights
    const size_t WPK_ELEMS = (size_t)NLAYER * NTILES * KTILES * TILE_ELEMS;
    __bf16* actBase = (__bf16*)(ws + WPK_ELEMS * sizeof(__bf16)); // 3 x 16 MB activation ring
    const size_t ACT_ELEMS = (size_t)BROWS * DD;
    __bf16* act[3] = { actBase, actBase + ACT_ELEMS, actBase + 2 * ACT_ELEMS };

    // 1) pack/convert all weights (one launch, coalesced both sides via LDS)
    pack_w_kernel<<<NLAYER * 64 * 16, 256, 0, stream>>>(W, Wpk);
    // 2) x -> bf16 into ring slot 2 (acts as "out[-1]")
    cvt_f32_bf16_kernel<<<(BROWS * DD / 8) / 256, 256, 0, stream>>>(x, act[2]);

    // 3) 16 dependent GEMM layers
    dim3 grid(DD / 128, BROWS / 128);   // (16, 32)
    for (int l = 0; l < NLAYER; ++l) {
        const __bf16* A1 = act[(l + 2) % 3];                      // y_{l-1} (or x for l==0)
        const __bf16* A2 = (l >= 2) ? act[(l + 1) % 3] : nullptr; // y_{l-2}
        __bf16* ob = (l < NLAYER - 1) ? act[l % 3] : nullptr;
        float*  of = (l == NLAYER - 1) ? out : nullptr;
        layer_kernel<<<grid, 256, 0, stream>>>(
            A1, A2, Wpk + (size_t)l * NTILES * KTILES * TILE_ELEMS,
            bias + (size_t)l * DD, ob, of);
    }
}
